// MultiHeadAttention_64312840290635
// MI455X (gfx1250) — matrix operand order repaired and checked by reading, not compiled, not run on hardware
//
#include <hip/hip_runtime.h>
#include <hip/hip_bf16.h>
#include <stdint.h>

#define D_MODEL 1024
#define NHEADS  16
#define HDIM    64
#define BATCH   2
#define SEQ     2048
#define ROWS    (BATCH*SEQ)      /* 4096 */
#define QKV_N   (3*D_MODEL)      /* 3072 */

typedef __attribute__((ext_vector_type(16))) __bf16 v16bf;
typedef __attribute__((ext_vector_type(8)))  float  v8f;

union Frag { v16bf h; uint32_t u[8]; };

__device__ __forceinline__ uint16_t f2bf(float x){
  uint32_t u = __builtin_bit_cast(uint32_t, x);
  return (uint16_t)((u + 0x7FFFu + ((u >> 16) & 1u)) >> 16);
}
__device__ __forceinline__ uint32_t pack2(float lo, float hi){
  return (uint32_t)f2bf(lo) | ((uint32_t)f2bf(hi) << 16);
}
__device__ __forceinline__ v8f wmma_bf16(const Frag& a, const Frag& b, v8f c){
  return __builtin_amdgcn_wmma_f32_16x16x32_bf16(false, a.h, false, b.h, (short)0, c, false, false);
}
// A-fragment (16x32 bf16, lane = M row): VGPR j <- K pair 16*half + 2j
__device__ __forceinline__ void load_afrag(Frag& f, const uint16_t* rowptr, int kb, int half){
  #pragma unroll
  for (int j = 0; j < 8; ++j){
    int k0 = kb + 16*half + 2*j;
    f.u[j] = *(const uint32_t*)(rowptr + k0);
  }
}
// B-fragment (32x16 bf16, lane = N col): VGPR j <- K pair 16*half + 2j
__device__ __forceinline__ void load_bfrag(Frag& f, const uint16_t* colptr, int kb, int half){
  #pragma unroll
  for (int j = 0; j < 8; ++j){
    int kk = kb + 16*half + 2*j;
    f.u[j] = *(const uint32_t*)(colptr + kk);
  }
}

// ---------------- conversion kernels ----------------
__global__ void cvt_f32_to_bf16_x4(const float4* __restrict__ in, uint2* __restrict__ out, int n4){
  int i = blockIdx.x * blockDim.x + threadIdx.x;
  if (i < n4) {
    float4 v = in[i];
    uint2 r; r.x = pack2(v.x, v.y); r.y = pack2(v.z, v.w);
    out[i] = r;
  }
}
// W (K x N) fp32 row-major -> WT (N x K) bf16 row-major, 32x32 LDS-tiled
__global__ void transpose_to_bf16(const float* __restrict__ W, uint16_t* __restrict__ WT, int K, int N){
  __shared__ float tile[32][33];
  int ntiles_n = N >> 5;
  int bx = blockIdx.x % ntiles_n;     // n tile
  int by = blockIdx.x / ntiles_n;     // k tile
  int tx = threadIdx.x & 31;
  int ty = threadIdx.x >> 5;          // 0..7
  #pragma unroll
  for (int i = 0; i < 4; ++i)
    tile[ty + i*8][tx] = W[(size_t)(by*32 + ty + i*8) * N + bx*32 + tx];
  __syncthreads();
  #pragma unroll
  for (int i = 0; i < 4; ++i)
    WT[(size_t)(bx*32 + ty + i*8) * K + by*32 + tx] = f2bf(tile[tx][ty + i*8]);
}

// ---- shared GEMM core: 16x64 tile / wave, K fully unrolled ----
// Full unroll removes all loop-carried register constraints: no rotation copies,
// and the scheduler hoists chunk i+1 loads above chunk i WMMAs (partial waits).
__device__ __forceinline__ void gemm_16x64(const uint16_t* __restrict__ Arow,
                                           const uint16_t* const Bp[4],
                                           int half, v8f acc[4]){
  #pragma unroll
  for (int kb = 0; kb < D_MODEL; kb += 32) {
    Frag a;
    load_afrag(a, Arow, kb, half);
    Frag b[4];
    #pragma unroll
    for (int d = 0; d < 4; ++d) load_bfrag(b[d], Bp[d], kb, half);
    #pragma unroll
    for (int d = 0; d < 4; ++d) acc[d] = wmma_bf16(a, b[d], acc[d]);
  }
}

// ---------------- QKV projection ----------------
__global__ void qkv_gemm(const uint16_t* __restrict__ Xb, const uint16_t* __restrict__ WT,
                         const float* __restrict__ bqkv,
                         uint16_t* __restrict__ Q, uint16_t* __restrict__ K,
                         uint16_t* __restrict__ VT){
  int wave = (blockIdx.x * blockDim.x + threadIdx.x) >> 5;
  int lane = threadIdx.x & 31;
  const int NT = QKV_N / 64;               // 48
  int nt = wave % NT;
  int mt = wave / NT;                      // < 256
  int nlo = lane & 15, half = lane >> 4;

  const uint16_t* Arow = Xb + (size_t)(mt*16 + nlo) * D_MODEL;
  const uint16_t* Bp[4];
  #pragma unroll
  for (int d = 0; d < 4; ++d)
    Bp[d] = WT + (size_t)(nt*64 + d*16 + nlo) * D_MODEL;

  v8f z = {0.f,0.f,0.f,0.f,0.f,0.f,0.f,0.f};
  v8f acc[4]; acc[0]=z; acc[1]=z; acc[2]=z; acc[3]=z;
  gemm_16x64(Arow, Bp, half, acc);

  #pragma unroll
  for (int d = 0; d < 4; ++d) {
    int col = nt * 64 + d * 16 + nlo;
    int sec = col >> 10;                   // 0=Q 1=K 2=V
    int cw  = col & 1023;
    int h   = cw >> 6;
    int dh  = cw & 63;
    float bias = bqkv[col];
    #pragma unroll
    for (int r = 0; r < 8; ++r) {
      int row = mt*16 + r + 8*half;        // global (b*S + s)
      int b = row >> 11, s = row & 2047;
      int bh = b * NHEADS + h;
      float v = acc[d][r] + bias;
      if (sec == 0) {
        v *= 0.125f;                        // 1/sqrt(HDIM) folded into Q
        Q[((size_t)bh * SEQ + s) * HDIM + dh] = f2bf(v);
      } else if (sec == 1) {
        K[((size_t)bh * SEQ + s) * HDIM + dh] = f2bf(v);
      } else {
        VT[((size_t)bh * HDIM + dh) * SEQ + s] = f2bf(v);  // V stored transposed
      }
    }
  }
}

// ---------------- flash attention: one wave per 16-query tile ----------------
__global__ void flash_attn(const uint16_t* __restrict__ Qb, const uint16_t* __restrict__ Kb,
                           const uint16_t* __restrict__ Vt, uint16_t* __restrict__ Xo){
  __shared__ uint16_t Plds[4][16 * 32];
  int wave = (blockIdx.x * blockDim.x + threadIdx.x) >> 5;
  int lane = threadIdx.x & 31;
  int wib  = (threadIdx.x >> 5) & 3;
  int qt = wave & 127;                     // S/16 = 128 tiles
  int bh = wave >> 7;                      // 32 (b,h) pairs
  int nlo = lane & 15, half = lane >> 4;
  const uint16_t* Qh = Qb + (size_t)bh * SEQ * HDIM;
  const uint16_t* Kh = Kb + (size_t)bh * SEQ * HDIM;
  const uint16_t* Vh = Vt + (size_t)bh * HDIM * SEQ;
  uint16_t* P = &Plds[wib][0];
  int q0 = qt * 16;

  // Q^T B-fragments for the two 32-wide Dh chunks
  Frag qf[2];
  #pragma unroll
  for (int c = 0; c < 2; ++c)
    load_bfrag(qf[c], Qh + (size_t)(q0 + nlo) * HDIM, c*32, half);

  v8f z = {0.f,0.f,0.f,0.f,0.f,0.f,0.f,0.f};
  v8f o[4]; o[0]=z; o[1]=z; o[2]=z; o[3]=z;
  float l = 0.f, mrow = -1e30f;
  int kend = (q0 + 16 + 31) & ~31;

  for (int kb = 0; kb < kend; kb += 32) {
    // V B-fragments issued first: address depends only on kb, overlaps everything below
    Frag vf[4];
    #pragma unroll
    for (int d = 0; d < 4; ++d)
      load_bfrag(vf[d], Vh + (size_t)(d*16 + nlo) * SEQ, kb, half);

    // S^T = K_tile (16k x 64d) @ Q^T (64d x 16q): query lives on the lane axis
    v8f st[2]; st[0]=z; st[1]=z;
    #pragma unroll
    for (int sub = 0; sub < 2; ++sub) {
      const uint16_t* krow = Kh + (size_t)(kb + sub*16 + nlo) * HDIM;
      #pragma unroll
      for (int c = 0; c < 2; ++c) {
        Frag kf;
        load_afrag(kf, krow, c*32, half);
        st[sub] = wmma_bf16(kf, qf[c], st[sub]);
      }
    }
    // causal mask + online softmax (per-lane query stats)
    int qg = q0 + nlo;
    float mnew = mrow;
    #pragma unroll
    for (int sub = 0; sub < 2; ++sub)
      #pragma unroll
      for (int r = 0; r < 8; ++r) {
        int key = kb + sub*16 + r + 8*half;
        float s = st[sub][r];
        s = (key > qg) ? -1e30f : s;
        st[sub][r] = s;
        mnew = fmaxf(mnew, s);
      }
    mnew = fmaxf(mnew, __shfl_xor(mnew, 16, 32));
    float fac = exp2f((mrow - mnew) * 1.44269504f);
    float rsum = 0.f;
    #pragma unroll
    for (int sub = 0; sub < 2; ++sub)
      #pragma unroll
      for (int r = 0; r < 8; ++r) {
        float p = exp2f((st[sub][r] - mnew) * 1.44269504f);
        st[sub][r] = p;
        rsum += p;
      }
    rsum += __shfl_xor(rsum, 16, 32);
    l = l * fac + rsum;
    mrow = mnew;

    // P^T (C layout) -> LDS [query][key] bf16, then re-read as A-fragment
    #pragma unroll
    for (int sub = 0; sub < 2; ++sub)
      #pragma unroll
      for (int r = 0; r < 8; r += 2) {
        int key = sub*16 + r + 8*half;
        *(uint32_t*)(P + nlo*32 + key) = pack2(st[sub][r], st[sub][r+1]);
      }
    asm volatile("s_wait_dscnt 0" ::: "memory");
    Frag pf;
    load_afrag(pf, P + nlo*32, 0, half);

    // rescale O (query = r + 8*half on the VGPR axis -> broadcast fac via shfl)
    #pragma unroll
    for (int r = 0; r < 8; ++r) {
      float f = __shfl(fac, r + 8*half, 32);
      #pragma unroll
      for (int d = 0; d < 4; ++d) o[d][r] *= f;
    }
    // O += P (16q x 32k) @ V (32k x 64d)
    #pragma unroll
    for (int d = 0; d < 4; ++d)
      o[d] = wmma_bf16(pf, vf[d], o[d]);
  }

  float linv = 1.0f / l;
  int b = bh >> 4, h = bh & 15;
  #pragma unroll
  for (int r = 0; r < 8; ++r) {
    float li = __shfl(linv, r + 8*half, 32);
    int s = q0 + r + 8*half;
    size_t base = ((size_t)b * SEQ + s) * D_MODEL + h * HDIM;
    #pragma unroll
    for (int d = 0; d < 4; ++d)
      Xo[base + d*16 + nlo] = f2bf(o[d][r] * li);
  }
}

// ---------------- output projection ----------------
__global__ void out_gemm(const uint16_t* __restrict__ Xo, const uint16_t* __restrict__ WT,
                         const float* __restrict__ bout, float* __restrict__ out){
  int wave = (blockIdx.x * blockDim.x + threadIdx.x) >> 5;
  int lane = threadIdx.x & 31;
  const int NT = D_MODEL / 64;             // 16
  int nt = wave % NT;
  int mt = wave / NT;                      // < 256
  int nlo = lane & 15, half = lane >> 4;

  const uint16_t* Arow = Xo + (size_t)(mt*16 + nlo) * D_MODEL;
  const uint16_t* Bp[4];
  #pragma unroll
  for (int d = 0; d < 4; ++d)
    Bp[d] = WT + (size_t)(nt*64 + d*16 + nlo) * D_MODEL;

  v8f z = {0.f,0.f,0.f,0.f,0.f,0.f,0.f,0.f};
  v8f acc[4]; acc[0]=z; acc[1]=z; acc[2]=z; acc[3]=z;
  gemm_16x64(Arow, Bp, half, acc);

  #pragma unroll
  for (int d = 0; d < 4; ++d) {
    int col = nt * 64 + d * 16 + nlo;
    float bias = bout[col];
    #pragma unroll
    for (int r = 0; r < 8; ++r) {
      int row = mt*16 + r + 8*half;
      out[(size_t)row * D_MODEL + col] = acc[d][r] + bias;
    }
  }
}

extern "C" void kernel_launch(void* const* d_in, const int* in_sizes, int n_in,
                              void* d_out, int out_size, void* d_ws, size_t ws_size,
                              hipStream_t stream) {
  (void)in_sizes; (void)n_in; (void)out_size; (void)ws_size;
  const float* x    = (const float*)d_in[0];
  const float* Wqkv = (const float*)d_in[1];
  const float* bqkv = (const float*)d_in[2];
  const float* Wout = (const float*)d_in[3];
  const float* bout = (const float*)d_in[4];
  float* out = (float*)d_out;

  char* p = (char*)d_ws;
  uint16_t* Xb    = (uint16_t*)p; p += (size_t)ROWS * D_MODEL * 2;
  uint16_t* WqkvT = (uint16_t*)p; p += (size_t)QKV_N * D_MODEL * 2;
  uint16_t* WoutT = (uint16_t*)p; p += (size_t)D_MODEL * D_MODEL * 2;
  uint16_t* Qb    = (uint16_t*)p; p += (size_t)BATCH * NHEADS * SEQ * HDIM * 2;
  uint16_t* Kb    = (uint16_t*)p; p += (size_t)BATCH * NHEADS * SEQ * HDIM * 2;
  uint16_t* Vt    = (uint16_t*)p; p += (size_t)BATCH * NHEADS * SEQ * HDIM * 2;
  uint16_t* Xo    = (uint16_t*)p; p += (size_t)ROWS * D_MODEL * 2;

  {
    int n4 = (ROWS * D_MODEL) / 4;
    cvt_f32_to_bf16_x4<<<(n4 + 255) / 256, 256, 0, stream>>>((const float4*)x, (uint2*)Xb, n4);
  }
  transpose_to_bf16<<<(QKV_N/32) * (D_MODEL/32), 256, 0, stream>>>(Wqkv, WqkvT, D_MODEL, QKV_N);
  transpose_to_bf16<<<(D_MODEL/32) * (D_MODEL/32), 256, 0, stream>>>(Wout, WoutT, D_MODEL, D_MODEL);
  // 256 M-tiles x 48 N64-tiles = 12288 waves, 4 waves/block
  qkv_gemm<<<(ROWS / 16) * (QKV_N / 64) / 4, 128, 0, stream>>>(Xb, WqkvT, bqkv, Qb, Kb, Vt);
  // 32 (b,h) x 128 query tiles = 4096 waves
  flash_attn<<<(BATCH * NHEADS * (SEQ / 16)) / 4, 128, 0, stream>>>(Qb, Kb, Vt, Xo);
  // 256 x 16 = 4096 waves
  out_gemm<<<(ROWS / 16) * (D_MODEL / 64) / 4, 128, 0, stream>>>(Xo, WoutT, bout, out);
}
